// _BNBLinear4bit_81406810128701
// MI455X (gfx1250) — matrix operand order repaired and checked by reading, not compiled, not run on hardware
//
#include <hip/hip_runtime.h>

// ---------------------------------------------------------------------------
// NF4 (bitsandbytes) 4-bit quantized linear for MI455X / gfx1250.
//   Stage 1a: dequant W codes -> bf16   (ws, 32 MB)
//   Stage 1b: cast x fp32 -> bf16       (ws, 64 MB)
//   Stage 2 : bf16 WMMA GEMM, 128x256 workgroup tile, 64x64 per wave
//             (16 v_wmma per K-step per wave, wmma:ds-load = 1.0),
//             double-buffered LDS filled by the Tensor Data Mover
//             (tensor_load_to_lds, TENSORcnt) with hardware row padding,
//             + bias -> fp32 out.
// Both bf16 operands (96 MB) fit in the 192 MB L2, so tile re-reads during
// the GEMM never touch HBM.
// ---------------------------------------------------------------------------

typedef __attribute__((ext_vector_type(16))) __bf16 v16bf;
typedef __attribute__((ext_vector_type(8)))  __bf16 v8bf;
typedef __attribute__((ext_vector_type(8)))  float  v8f;
typedef __attribute__((ext_vector_type(4)))  int    v4i;
typedef __attribute__((ext_vector_type(8)))  int    v8i;
typedef __attribute__((ext_vector_type(4)))  unsigned v4u;
typedef __attribute__((ext_vector_type(4)))  float  v4f;

#define I_DIM 4096
#define O_DIM 4096
#define M_DIM 8192   // B * S = 4 * 2048
#define K_DIM 4096
#define QBLK  64
#define BM 128
#define BN 256
#define BK 64
#define LDT 72       // padded LDS stride (elements); 144 B row pitch

__constant__ float NF4_TBL[16] = {
    -1.0f, -0.6961928009986877f, -0.5250730514526367f, -0.39491748809814453f,
    -0.28444138169288635f, -0.18477343022823334f, -0.09105003625154495f, 0.0f,
    0.07958029955625534f, 0.16093020141124725f, 0.24611230194568634f,
    0.33791524171829224f, 0.44070982933044434f, 0.5626170039176941f,
    0.7229568362236023f, 1.0f};

// ---------------- Stage 1a: NF4 + double-quant dequantization --------------
// One thread produces 8 consecutive bf16 weights (one aligned 16B store).
__global__ __launch_bounds__(256)
void dequant_w_kernel(const int* __restrict__ w_codes,
                      const int* __restrict__ absmax_codes,
                      const float* __restrict__ absmax_scale,
                      const float* __restrict__ absmax_offset,
                      __bf16* __restrict__ w_bf16) {
  __shared__ float lut[16];
  if (threadIdx.x < 16) lut[threadIdx.x] = NF4_TBL[threadIdx.x];
  __syncthreads();

  const int g  = blockIdx.x * blockDim.x + threadIdx.x;  // 8 elems per thread
  const int o  = g >> 9;                                 // I/8 = 512 chunks/row
  const int i0 = (g & 511) << 3;
  const int blk = o * (I_DIM / QBLK) + (i0 >> 6);        // per-64 block index
  const float am = (float)absmax_codes[blk] * (1.0f / 255.0f) *
                   absmax_scale[blk >> 8] + absmax_offset[0];

  const size_t base = (size_t)o * I_DIM + i0;
  const v4i c0 = *(const v4i*)(w_codes + base);
  const v4i c1 = *(const v4i*)(w_codes + base + 4);

  v8bf w;
  w[0] = (__bf16)(lut[c0[0] & 15] * am);
  w[1] = (__bf16)(lut[c0[1] & 15] * am);
  w[2] = (__bf16)(lut[c0[2] & 15] * am);
  w[3] = (__bf16)(lut[c0[3] & 15] * am);
  w[4] = (__bf16)(lut[c1[0] & 15] * am);
  w[5] = (__bf16)(lut[c1[1] & 15] * am);
  w[6] = (__bf16)(lut[c1[2] & 15] * am);
  w[7] = (__bf16)(lut[c1[3] & 15] * am);
  *(v8bf*)(w_bf16 + base) = w;
}

// ---------------- Stage 1b: x fp32 -> bf16 ---------------------------------
__global__ __launch_bounds__(256)
void cast_x_kernel(const float* __restrict__ x, __bf16* __restrict__ xb) {
  const size_t g = (size_t)blockIdx.x * blockDim.x + threadIdx.x;
  const v4f a = *(const v4f*)(x + g * 8);
  const v4f b = *(const v4f*)(x + g * 8 + 4);
  v8bf o;
  o[0] = (__bf16)a[0]; o[1] = (__bf16)a[1];
  o[2] = (__bf16)a[2]; o[3] = (__bf16)a[3];
  o[4] = (__bf16)b[0]; o[5] = (__bf16)b[1];
  o[6] = (__bf16)b[2]; o[7] = (__bf16)b[3];
  *(v8bf*)(xb + g * 8) = o;
}

// ---------------- Stage 2: bf16 WMMA GEMM + bias ---------------------------
// A: [M,K] bf16 row-major (x).  B: [N,K] bf16 row-major (w[o][i] -> lane=N
// column layout for the WMMA B operand, no transpose needed).
static __device__ inline v16bf mk_frag(v8bf lo, v8bf hi) {
  v16bf r;
#pragma unroll
  for (int i = 0; i < 8; ++i) { r[i] = lo[i]; r[i + 8] = hi[i]; }
  return r;
}

// TDM: DMA a 2D bf16 tile (tile_d1 rows x tile_d0 cols) from a row-major
// tensor (row stride = stride0 elements) into LDS at lds_addr, inserting
// 16 B of padding after every 128 B row (pad_interval=4 -> 32 DWORDs,
// pad_amount=3 -> 4 DWORDs) => LDS row pitch 144 B == LDT*2.
static __device__ inline void tdm_load_2d(unsigned lds_addr, const void* gaddr,
                                          unsigned tensor_d0, unsigned tensor_d1,
                                          unsigned tile_d0, unsigned tile_d1,
                                          unsigned stride0) {
  const unsigned long long ga = (unsigned long long)(uintptr_t)gaddr;
  v4u g0;
  g0[0] = 1u;                                   // count=1, user descriptor
  g0[1] = lds_addr;                             // bits 63:32  lds_addr
  g0[2] = (unsigned)ga;                         // bits 95:64  global_addr lo
  g0[3] = (unsigned)((ga >> 32) & 0x1FFFFFFu)   // bits 120:96 global_addr hi
          | (2u << 30);                         // bits 127:126 type=2 (image)
  v8i g1;
  g1[0] = (int)((1u << 16)      // data_size = 1 -> 2 bytes/element
                | (1u << 20)    // pad_enable
                | (4u << 22)    // pad_interval: 32 DWORDs (128 B)
                | (3u << 25));  // pad_amount:   4 DWORDs (16 B)
  g1[1] = (int)(tensor_d0 << 16);                       // dim0[15:0]
  g1[2] = (int)((tensor_d0 >> 16) | (tensor_d1 << 16)); // dim0 hi | dim1 lo
  g1[3] = (int)((tensor_d1 >> 16) | (tile_d0 << 16));   // dim1 hi | tile_d0
  g1[4] = (int)tile_d1;                                 // tile_d1 | tile_d2=0
  g1[5] = (int)stride0;                                 // dim0_stride lo
  g1[6] = 0;                                            // stride hi | dim1_stride
  g1[7] = 0;
  const v4i gz4 = {0, 0, 0, 0};                         // groups 2/3 unused (2D)
  const v8i gz8 = {0, 0, 0, 0, 0, 0, 0, 0};
  __builtin_amdgcn_tensor_load_to_lds(g0, g1, gz4, gz4, gz8, 0);
}

__global__ __launch_bounds__(256)
void gemm_bf16_kernel(const __bf16* __restrict__ A, const __bf16* __restrict__ B,
                      const float* __restrict__ bias, float* __restrict__ C) {
  __shared__ __align__(16) __bf16 As[2][BM * LDT];
  __shared__ __align__(16) __bf16 Bs[2][BN * LDT];

  const int tid   = threadIdx.x;
  const int lane  = tid & 31;
  const int wave  = tid >> 5;
  const int wm    = (wave >> 2) * 64;   // wave M offset in tile (2 wave-rows)
  const int wn    = (wave & 3) * 64;    // wave N offset in tile (4 wave-cols)
  const int lhalf = lane >> 4;          // 0 | 1
  const int lmod  = lane & 15;
  const int m0    = blockIdx.y * BM;
  const int n0    = blockIdx.x * BN;

  // One wave drives the Tensor Data Mover for the whole workgroup: two
  // descriptor issues per K-tile (A: 128x64, B: 256x64), TENSORcnt-tracked.
  auto issue_tile = [&](int buf, int k0) {
    if (wave == 0) {
      tdm_load_2d((unsigned)(uintptr_t)&As[buf][0],
                  &A[(size_t)m0 * K_DIM + k0],
                  K_DIM, M_DIM, BK, BM, K_DIM);
      tdm_load_2d((unsigned)(uintptr_t)&Bs[buf][0],
                  &B[(size_t)n0 * K_DIM + k0],
                  K_DIM, O_DIM, BK, BN, K_DIM);
    }
  };
  auto publish_tile = [&]() {
    if (wave == 0) __builtin_amdgcn_s_wait_tensorcnt(0);
    __syncthreads();
  };

  const v8f vzero = {0.f, 0.f, 0.f, 0.f, 0.f, 0.f, 0.f, 0.f};
  v8f acc[4][4];
#pragma unroll
  for (int mf = 0; mf < 4; ++mf)
#pragma unroll
    for (int nf = 0; nf < 4; ++nf) acc[mf][nf] = vzero;

  // prologue: fill buffer 0
  issue_tile(0, 0);
  publish_tile();

  for (int k0 = 0; k0 < K_DIM; k0 += BK) {
    const int cur = (k0 >> 6) & 1;
    // DMA the next K-tile into the other LDS buffer while we compute.
    // buf[cur^1] was last read in the previous iteration; the barrier that
    // ended it guarantees every wave is done with those reads.
    if (k0 + BK < K_DIM) issue_tile(cur ^ 1, k0 + BK);

#pragma unroll
    for (int kk = 0; kk < BK; kk += 32) {
      v16bf afrag[4];
#pragma unroll
      for (int mf = 0; mf < 4; ++mf) {
        // A 16x32 bf16 layout: lanes 0-15 hold K 0..7 (v0-3) & 16..23 (v4-7),
        // lanes 16-31 hold K 8..15 & 24..31.
        const __bf16* p =
            &As[cur][(wm + mf * 16 + lmod) * LDT + kk + lhalf * 8];
        afrag[mf] = mk_frag(*(const v8bf*)p, *(const v8bf*)(p + 16));
      }
      v16bf bfrag[4];
#pragma unroll
      for (int nf = 0; nf < 4; ++nf) {
        // B 32x16 bf16 layout: lanes 0-15 hold K 0..7 & 16..23, lanes 16-31 K 8..15 & 24..31.
        const __bf16* p =
            &Bs[cur][(wn + nf * 16 + lmod) * LDT + kk + lhalf * 8];
        bfrag[nf] = mk_frag(*(const v8bf*)p, *(const v8bf*)(p + 16));
      }
#pragma unroll
      for (int mf = 0; mf < 4; ++mf)
#pragma unroll
        for (int nf = 0; nf < 4; ++nf)
          acc[mf][nf] = __builtin_amdgcn_wmma_f32_16x16x32_bf16(
              false, afrag[mf], false, bfrag[nf], (short)0, acc[mf][nf],
              false, false);
    }

    // publish the DMA'd buffer: wave 0 waits TENSORcnt==0, then the
    // block-wide barrier makes the new tile visible to every wave.
    publish_tile();
  }

  // ---- epilogue: bias + store ----
  float bv[4];
#pragma unroll
  for (int nf = 0; nf < 4; ++nf) bv[nf] = bias[n0 + wn + nf * 16 + lmod];
#pragma unroll
  for (int mf = 0; mf < 4; ++mf) {
#pragma unroll
    for (int nf = 0; nf < 4; ++nf) {
      const int ncol = n0 + wn + nf * 16 + lmod;
#pragma unroll
      for (int r = 0; r < 8; ++r) {
        // C 16x16 f32 layout: vgpr r, lanes 0-15 -> M=r, lanes 16-31 -> M=8+r
        const int mrow = m0 + wm + mf * 16 + lhalf * 8 + r;
        C[(size_t)mrow * O_DIM + ncol] = acc[mf][nf][r] + bv[nf];
      }
    }
  }
}

// ---------------------------------------------------------------------------
extern "C" void kernel_launch(void* const* d_in, const int* in_sizes, int n_in,
                              void* d_out, int out_size, void* d_ws, size_t ws_size,
                              hipStream_t stream) {
  const float* x             = (const float*)d_in[0];
  const int*   w_codes       = (const int*)d_in[1];
  const int*   absmax_codes  = (const int*)d_in[2];
  const float* absmax_scale  = (const float*)d_in[3];
  const float* absmax_offset = (const float*)d_in[4];
  const float* bias          = (const float*)d_in[5];
  float*       out           = (float*)d_out;

  __bf16* xb = (__bf16*)d_ws;                                      // 64 MB
  __bf16* wb = (__bf16*)((char*)d_ws + (size_t)M_DIM * K_DIM * 2); // +32 MB

  dequant_w_kernel<<<(O_DIM * I_DIM / 8) / 256, 256, 0, stream>>>(
      w_codes, absmax_codes, absmax_scale, absmax_offset, wb);
  cast_x_kernel<<<(M_DIM * K_DIM / 8) / 256, 256, 0, stream>>>(x, xb);

  dim3 grid(O_DIM / BN, M_DIM / BM);
  gemm_bf16_kernel<<<grid, 256, 0, stream>>>(xb, wb, bias, out);
}
